// RoutingAssignment_70265664963132
// MI455X (gfx1250) — matrix operand order repaired and checked by reading, not compiled, not run on hardware
//
#include <hip/hip_runtime.h>
#include <hip/hip_bf16.h>

// ---------------------------------------------------------------------------
// Problem constants (from reference): x[4,4096,128], adj[4,4096,4096],
// W[128,512], b[512]; output softmax(s) [4,4096,512] fp32.
// ---------------------------------------------------------------------------
constexpr int BB = 4;       // batch
constexpr int NN = 4096;    // nodes
constexpr int FF = 128;     // features
constexpr int KC = 512;     // capsules / NNEXT
constexpr int RTN_ITERS = 3;

typedef __attribute__((ext_vector_type(16))) __bf16 v16bf;
typedef __attribute__((ext_vector_type(8)))  float  v8f;

#if __has_builtin(__builtin_amdgcn_global_load_async_to_lds_b128) && \
    __has_builtin(__builtin_amdgcn_s_wait_asynccnt)
#define USE_ASYNC 1
typedef int nvi4 __attribute__((vector_size(16)));       // native int4 vector
typedef __attribute__((address_space(1))) nvi4 gl_i4;    // global int4
typedef __attribute__((address_space(3))) nvi4 lds_i4;   // LDS int4
#else
#define USE_ASYNC 0
#endif

union Frag16 {
    uint4          q[2];
    v16bf          v;
    unsigned short s16[16];
};

__device__ __forceinline__ unsigned short f2bf(float f) {
    unsigned int u = __float_as_uint(f);
    u += 0x7FFFu + ((u >> 16) & 1u);       // round-to-nearest-even
    return (unsigned short)(u >> 16);
}

// ---------------------------------------------------------------------------
// Block reductions (256 threads, wave32)
// ---------------------------------------------------------------------------
__device__ __forceinline__ float warp_sum(float v) {
#pragma unroll
    for (int o = 16; o > 0; o >>= 1) v += __shfl_xor(v, o, 32);
    return v;
}
__device__ __forceinline__ float warp_max(float v) {
#pragma unroll
    for (int o = 16; o > 0; o >>= 1) v = fmaxf(v, __shfl_xor(v, o, 32));
    return v;
}
__device__ __forceinline__ float block_sum(float v, float* red) {
    v = warp_sum(v);
    __syncthreads();
    if ((threadIdx.x & 31) == 0) red[threadIdx.x >> 5] = v;
    __syncthreads();
    float t = red[0];
#pragma unroll
    for (int i = 1; i < 8; ++i) t += red[i];
    return t;
}
__device__ __forceinline__ float block_max(float v, float* red) {
    v = warp_max(v);
    __syncthreads();
    if ((threadIdx.x & 31) == 0) red[threadIdx.x >> 5] = v;
    __syncthreads();
    float t = red[0];
#pragma unroll
    for (int i = 1; i < 8; ++i) t = fmaxf(t, red[i]);
    return t;
}

// ---------------------------------------------------------------------------
// Generic bf16 WMMA GEMM:  C[M,Nc] (f32 or bf16) = A[M,K] * B[K,Nc] (+C) (+bias)
//  - A fp32 (converted on the fly, register-staged + prefetch) or bf16
//    (async DMA cache->LDS when available), row-major, leading dim lda
//  - B bf16 row-major ldb, transposed into LDS via VGPR staging
//  - block tile 128x128, 256 threads = 8 waves; wave w owns rows [w*16,w*16+16)
//  - double-buffered LDS: global traffic for tile k+1 overlaps WMMA on tile k
//  - K must be a multiple of 32
// ---------------------------------------------------------------------------
template <bool A_F32, bool ACCUM, bool BIAS, bool OUT_BF16>
__global__ __launch_bounds__(256) void wmma_gemm(
    const void* __restrict__ Ap, const unsigned short* __restrict__ Bp,
    const float* __restrict__ bias, void* __restrict__ Cp,
    int M, int Nc, int K, int lda, int ldb, int ldc,
    long long strideA, long long strideB, long long strideC) {
    __shared__ __attribute__((aligned(16))) unsigned short As[2][128 * 40];
    __shared__ __attribute__((aligned(16))) unsigned short Bs[2][128 * 40];

    const int tid  = threadIdx.x;
    const int wave = tid >> 5;
    const int lane = tid & 31;
    const int row0 = blockIdx.x * 128;
    const int col0 = blockIdx.y * 128;
    const long long z = blockIdx.z;

    const float*          Af  = (const float*)Ap + z * strideA;
    const unsigned short* Abf = (const unsigned short*)Ap + z * strideA;
    const unsigned short* Bm  = Bp + z * strideB;
    float*                Cf  = (float*)Cp + z * strideC;
    unsigned short*       Co  = (unsigned short*)Cp + z * strideC;

    v8f acc[8];
    if (ACCUM) {
#pragma unroll
        for (int ct = 0; ct < 8; ++ct)
#pragma unroll
            for (int r = 0; r < 8; ++r) {
                int m = row0 + wave * 16 + r + ((lane >> 4) << 3);
                int n = col0 + ct * 16 + (lane & 15);
                acc[ct][r] = Cf[(size_t)m * ldc + n];
            }
    } else {
#pragma unroll
        for (int ct = 0; ct < 8; ++ct) acc[ct] = (v8f){0.f,0.f,0.f,0.f,0.f,0.f,0.f,0.f};
    }

    float4 areg[4];      // f32-A staging
    uint4  aregu[2];     // bf16-A staging (fallback path)
    uint4  breg[2];      // B staging

    // issue global traffic for tile (k0) into registers / async DMA to LDS[buf]
    auto issue_tile = [&](int k0, int buf) {
        if (A_F32) {
#pragma unroll
            for (int p = 0; p < 4; ++p) {
                int r = p * 32 + (tid >> 3);
                int c = (tid & 7) * 4;
                const float* ga = Af + (size_t)(row0 + r) * lda + k0 + c;
                areg[p] = *(const float4*)ga;
                if (k0 + 64 < K) __builtin_prefetch(ga + 64, 0, 1);  // 2 tiles ahead
            }
        } else {
#if USE_ASYNC
#pragma unroll
            for (int p = 0; p < 2; ++p) {
                int r = p * 64 + (tid >> 2);
                int c = (tid & 3) * 8;
                __builtin_amdgcn_global_load_async_to_lds_b128(
                    (gl_i4*)(Abf + (size_t)(row0 + r) * lda + k0 + c),
                    (lds_i4*)&As[buf][r * 40 + c], 0, 0);
            }
#else
#pragma unroll
            for (int p = 0; p < 2; ++p) {
                int r = p * 64 + (tid >> 2);
                int c = (tid & 3) * 8;
                aregu[p] = *(const uint4*)(Abf + (size_t)(row0 + r) * lda + k0 + c);
            }
#endif
        }
#pragma unroll
        for (int p = 0; p < 2; ++p) {
            int kr = p * 16 + (tid >> 4);
            int nc = (tid & 15) * 8;
            breg[p] = *(const uint4*)(Bm + (size_t)(k0 + kr) * ldb + col0 + nc);
        }
    };

    // commit staged registers into LDS[buf] (+ wait async DMA)
    auto commit_tile = [&](int buf) {
        if (A_F32) {
#pragma unroll
            for (int p = 0; p < 4; ++p) {
                int r = p * 32 + (tid >> 3);
                int c = (tid & 7) * 4;
                unsigned short h4[4] = {f2bf(areg[p].x), f2bf(areg[p].y),
                                        f2bf(areg[p].z), f2bf(areg[p].w)};
                *(uint2*)&As[buf][r * 40 + c] = *(uint2*)h4;
            }
        } else {
#if !USE_ASYNC
#pragma unroll
            for (int p = 0; p < 2; ++p) {
                int r = p * 64 + (tid >> 2);
                int c = (tid & 3) * 8;
                *(uint4*)&As[buf][r * 40 + c] = aregu[p];
            }
#endif
        }
#pragma unroll
        for (int p = 0; p < 2; ++p) {
            int kr = p * 16 + (tid >> 4);
            int nc = (tid & 15) * 8;
            unsigned short e[8];
            *(uint4*)e = breg[p];
#pragma unroll
            for (int i = 0; i < 8; ++i) Bs[buf][(nc + i) * 40 + kr] = e[i];
        }
#if USE_ASYNC
        if (!A_F32) __builtin_amdgcn_s_wait_asynccnt(0);
#endif
    };

    const int ksteps = K >> 5;
    issue_tile(0, 0);
    commit_tile(0);
    int cur = 0;
    for (int i = 0; i < ksteps; ++i) {
        __syncthreads();                 // buf[cur] ready; buf[nxt] free
        const int nxt = cur ^ 1;
        const bool more = (i + 1) < ksteps;
        if (more) issue_tile((i + 1) * 32, nxt);   // overlaps WMMA below

        Frag16 afr;
        {
            int m  = wave * 16 + (lane & 15);
            int kh = (lane >> 4) * 8;
            afr.q[0] = *(const uint4*)&As[cur][m * 40 + kh];
            afr.q[1] = *(const uint4*)&As[cur][m * 40 + 16 + kh];
        }
#pragma unroll
        for (int ct = 0; ct < 8; ++ct) {
            Frag16 bfr;
            int n   = ct * 16 + (lane & 15);
            int kh2 = (lane >> 4) * 8;
            bfr.q[0] = *(const uint4*)&Bs[cur][n * 40 + kh2];
            bfr.q[1] = *(const uint4*)&Bs[cur][n * 40 + 16 + kh2];
            acc[ct] = __builtin_amdgcn_wmma_f32_16x16x32_bf16(
                false, afr.v, false, bfr.v, (short)0, acc[ct], false, false);
        }
        if (more) commit_tile(nxt);
        cur = nxt;
    }

    // ---- epilogue ----
#pragma unroll
    for (int ct = 0; ct < 8; ++ct) {
        float badd = BIAS ? bias[col0 + ct * 16 + (lane & 15)] : 0.f;
#pragma unroll
        for (int r = 0; r < 8; ++r) {
            int m = row0 + wave * 16 + r + ((lane >> 4) << 3);
            int n = col0 + ct * 16 + (lane & 15);
            float v = acc[ct][r] + badd;
            if (OUT_BF16) Co[(size_t)m * ldc + n] = f2bf(v);
            else          Cf[(size_t)m * ldc + n] = v;
        }
    }
}

// ---------------------------------------------------------------------------
// Elementwise / reduction kernels
// ---------------------------------------------------------------------------
__global__ void cvt_bf16_kernel(const float* __restrict__ in,
                                unsigned short* __restrict__ out, size_t n) {
    size_t i = (size_t)blockIdx.x * blockDim.x + threadIdx.x;
    if (i < n) out[i] = f2bf(in[i]);
}

__global__ void transpose_x_kernel(const float* __restrict__ x,
                                   unsigned short* __restrict__ xT) {
    // x[b][n][f] -> xT[b][f][n], bf16
    size_t i = (size_t)blockIdx.x * blockDim.x + threadIdx.x;
    if (i >= (size_t)BB * NN * FF) return;
    int f = (int)(i & (FF - 1));
    size_t nb = i >> 7;
    int n = (int)(nb & (NN - 1));
    int b = (int)(nb >> 12);
    xT[((size_t)b * FF + f) * NN + n] = f2bf(x[i]);
}

// L2-normalize(dim=2) + ReLU + BatchNorm over (B,K) per node; in-place on s.
__global__ __launch_bounds__(256) void norm_bn_kernel(float* __restrict__ s) {
    const int n   = blockIdx.x;
    const int tid = threadIdx.x;
    __shared__ float red[8];
    float vals[BB][2];
    float bnsum = 0.f, bnsq = 0.f;
#pragma unroll
    for (int b = 0; b < BB; ++b) {
        size_t base = ((size_t)b * NN + n) * KC;
        float v0 = s[base + tid];
        float v1 = s[base + tid + 256];
        float ss = block_sum(v0 * v0 + v1 * v1, red);
        float inv = 1.0f / fmaxf(sqrtf(ss), 1e-12f);
        v0 = fmaxf(v0 * inv, 0.f);
        v1 = fmaxf(v1 * inv, 0.f);
        vals[b][0] = v0; vals[b][1] = v1;
        bnsum += v0 + v1;
        bnsq  += v0 * v0 + v1 * v1;
    }
    float msum = block_sum(bnsum, red);
    float msq  = block_sum(bnsq, red);
    float mean = msum * (1.f / (BB * KC));
    float var  = msq * (1.f / (BB * KC)) - mean * mean;
    float sc   = rsqrtf(var + 1e-5f);
#pragma unroll
    for (int b = 0; b < BB; ++b) {
        size_t base = ((size_t)b * NN + n) * KC;
        s[base + tid]       = (vals[b][0] - mean) * sc;
        s[base + tid + 256] = (vals[b][1] - mean) * sc;
    }
}

template <bool OUT_BF16>
__global__ __launch_bounds__(256) void softmax_kernel(const float* __restrict__ s,
                                                      void* __restrict__ out) {
    size_t row = blockIdx.x;                 // (b*N + n)
    const float* sr = s + row * KC;
    int tid = threadIdx.x;
    __shared__ float red[8];
    float v0 = sr[tid], v1 = sr[tid + 256];
    float m  = block_max(fmaxf(v0, v1), red);
    float e0 = __expf(v0 - m), e1 = __expf(v1 - m);
    float inv = 1.f / block_sum(e0 + e1, red);
    if (OUT_BF16) {
        unsigned short* o = (unsigned short*)out + row * KC;
        o[tid] = f2bf(e0 * inv); o[tid + 256] = f2bf(e1 * inv);
    } else {
        float* o = (float*)out + row * KC;
        o[tid] = e0 * inv; o[tid + 256] = e1 * inv;
    }
}

// squash along f on transposed layout xnT[b][f][kc] (f32) -> bf16 same layout.
__global__ __launch_bounds__(512) void squash_kernel(const float* __restrict__ xnT,
                                                     unsigned short* __restrict__ xnTb) {
    int b = blockIdx.x;
    int kc = threadIdx.x;                    // 512 threads, coalesced columns
    const float* src = xnT + (size_t)b * FF * KC;
    float ss = 0.f;
#pragma unroll 4
    for (int f = 0; f < FF; ++f) {
        float v = src[(size_t)f * KC + kc];
        ss += v * v;
    }
    float scale = ss / (1.f + ss) / sqrtf(ss);   // mag_sq/(1+mag_sq) * 1/mag
    unsigned short* dst = xnTb + (size_t)b * FF * KC;
#pragma unroll 4
    for (int f = 0; f < FF; ++f)
        dst[(size_t)f * KC + kc] = f2bf(src[(size_t)f * KC + kc] * scale);
}

// ---------------------------------------------------------------------------
// Orchestration
// ---------------------------------------------------------------------------
extern "C" void kernel_launch(void* const* d_in, const int* in_sizes, int n_in,
                              void* d_out, int out_size, void* d_ws, size_t ws_size,
                              hipStream_t stream) {
    const float* x    = (const float*)d_in[0];   // [4,4096,128]
    const float* adj  = (const float*)d_in[1];   // [4,4096,4096]
    const float* W    = (const float*)d_in[2];   // [128,512]
    const float* bias = (const float*)d_in[3];   // [512]
    float* out = (float*)d_out;

    // workspace layout (bytes)
    char* ws = (char*)d_ws;
    const size_t SZ_XB  = (size_t)BB * NN * FF * 2;   // x   bf16 [b][n][f]
    const size_t SZ_XT  = (size_t)BB * FF * NN * 2;   // x^T bf16 [b][f][n]
    const size_t SZ_WB  = (size_t)FF * KC * 2;        // W   bf16
    const size_t SZ_H1  = (size_t)BB * NN * FF * 2;   // h1  bf16 [b][n][f]
    const size_t SZ_S   = (size_t)BB * NN * KC * 4;   // s   f32  [b][n][k]
    const size_t SZ_P   = (size_t)BB * NN * KC * 2;   // p   bf16 [b][n][k]
    const size_t SZ_XN  = (size_t)BB * FF * KC * 4;   // xnext^T f32 [b][f][k]
    unsigned short* xb   = (unsigned short*)(ws);
    unsigned short* xTb  = (unsigned short*)(ws + SZ_XB);
    unsigned short* Wb   = (unsigned short*)(ws + SZ_XB + SZ_XT);
    unsigned short* h1b  = (unsigned short*)(ws + SZ_XB + SZ_XT + SZ_WB);
    float*          s    = (float*)(ws + SZ_XB + SZ_XT + SZ_WB + SZ_H1);
    unsigned short* p    = (unsigned short*)(ws + SZ_XB + SZ_XT + SZ_WB + SZ_H1 + SZ_S);
    float*          xnT  = (float*)(ws + SZ_XB + SZ_XT + SZ_WB + SZ_H1 + SZ_S + SZ_P);
    unsigned short* xnTb = (unsigned short*)(ws + SZ_XB + SZ_XT + SZ_WB + SZ_H1 + SZ_S + SZ_P + SZ_XN);

    const size_t nX = (size_t)BB * NN * FF;
    cvt_bf16_kernel<<<(unsigned)((nX + 255) / 256), 256, 0, stream>>>(x, xb, nX);
    transpose_x_kernel<<<(unsigned)((nX + 255) / 256), 256, 0, stream>>>(x, xTb);
    const size_t nW = (size_t)FF * KC;
    cvt_bf16_kernel<<<(unsigned)((nW + 255) / 256), 256, 0, stream>>>(W, Wb, nW);

    // GEMM1: h1[b] = adj[b] (f32->bf16 on the fly) @ xb[b]; out bf16
    wmma_gemm<true, false, false, true><<<dim3(NN / 128, FF / 128, BB), 256, 0, stream>>>(
        (const void*)adj, xb, nullptr, (void*)h1b,
        NN, FF, NN, NN, FF, FF,
        (long long)NN * NN, (long long)NN * FF, (long long)NN * FF);

    // GEMM2: s = h1b @ Wb + bias  (flatten batch: M = B*N); out f32
    wmma_gemm<false, false, true, false><<<dim3((BB * NN) / 128, KC / 128, 1), 256, 0, stream>>>(
        (const void*)h1b, Wb, bias, (void*)s,
        BB * NN, KC, FF, FF, KC, KC, 0, 0, 0);

    // normalize + relu + batchnorm (in place on s)
    norm_bn_kernel<<<NN, 256, 0, stream>>>(s);

    for (int it = 0; it < RTN_ITERS; ++it) {
        // p = softmax(s) -> bf16
        softmax_kernel<true><<<BB * NN, 256, 0, stream>>>(s, (void*)p);
        // xnext^T[b] = xT[b] @ p[b]   (M=128, N=512, K=4096); out f32
        wmma_gemm<false, false, false, false><<<dim3(1, KC / 128, BB), 256, 0, stream>>>(
            (const void*)xTb, p, nullptr, (void*)xnT,
            FF, KC, NN, NN, KC, KC,
            (long long)FF * NN, (long long)NN * KC, (long long)FF * KC);
        // squash per capsule (column-wise on transposed layout) -> bf16
        squash_kernel<<<BB, 512, 0, stream>>>(xnT, xnTb);
        // s[b] += xb[b] @ xnTb[b]   (M=4096, N=512, K=128), C accumulate
        wmma_gemm<false, true, false, false><<<dim3(NN / 128, KC / 128, BB), 256, 0, stream>>>(
            (const void*)xb, xnTb, nullptr, (void*)s,
            NN, KC, FF, FF, KC, KC,
            (long long)NN * FF, (long long)FF * KC, (long long)NN * KC);
    }

    // final softmax -> f32 output
    softmax_kernel<false><<<BB * NN, 256, 0, stream>>>(s, (void*)out);
}
